// VectorQuantizer_62663572849177
// MI455X (gfx1250) — matrix operand order repaired and checked by reading, not compiled, not run on hardware
//
#include <hip/hip_runtime.h>

// VQ-VAE vector quantizer, fused WMMA distance-GEMM + argmin for MI455X (gfx1250).
//
// Inputs : d_in[0] = encoder_output f32 [32,256,32,32] (NCHW)
//          d_in[1] = codebook       f32 [1024,256]
// Output : d_out = [ vq_loss(1) | quantized_st(8388608, NCHW) | perplexity(1) |
//                    encodings(32768*1024 one-hot f32) ]
//
// Streaming outputs (encodings 134 MB, quantized 33.5 MB) use non-temporal
// stores so they don't evict the f16 codebook / x tiles from L2 (192 MB).

#define K_CODES   1024
#define D_EMB     256
#define HW        1024            // 32*32 spatial per image
#define N_ROWS    32768           // 32 * 32 * 32
#define XS_PITCH  264             // 256 + 8 halves pad -> conflict-free A gathers

#define QOFF   ((size_t)1)
#define PERPO  ((size_t)8388609)
#define EOFF   ((size_t)8388610)

typedef __attribute__((ext_vector_type(16))) _Float16 v16h;
typedef __attribute__((ext_vector_type(8)))  float    v8f;
typedef __attribute__((ext_vector_type(2)))  float    v2f;   // native vec for NT stores

// ---------------- kernel 1: codebook -> f16, norms, zero accumulators --------
__global__ __launch_bounds__(256) void vq_prep(const float* __restrict__ cb,
                                               _Float16* __restrict__ cb16,
                                               float* __restrict__ cnorm,
                                               int* __restrict__ counts,
                                               float* __restrict__ lossSum) {
  const int k = blockIdx.x;           // 0..1023 code rows
  const int t = threadIdx.x;          // 0..255  = d
  float v = cb[k * D_EMB + t];
  cb16[k * D_EMB + t] = (_Float16)v;
  float s = v * v;
  #pragma unroll
  for (int off = 16; off >= 1; off >>= 1) s += __shfl_xor(s, off, 32);
  __shared__ float red[8];
  if ((t & 31) == 0) red[t >> 5] = s;
  __syncthreads();
  if (t == 0) {
    float tot = 0.f;
    #pragma unroll
    for (int i = 0; i < 8; ++i) tot += red[i];
    cnorm[k]  = tot;
    counts[k] = 0;
    if (k == 0) *lossSum = 0.f;
  }
}

// ---------------- kernel 2: fused GEMM(argmin) + outputs ---------------------
__global__ __launch_bounds__(256) void vq_main(const float* __restrict__ enc,
                                               const float* __restrict__ cb,
                                               const _Float16* __restrict__ cb16,
                                               const float* __restrict__ cnorm,
                                               int* __restrict__ counts,
                                               float* __restrict__ lossSum,
                                               float* __restrict__ out) {
  __shared__ _Float16 Xs[16 * XS_PITCH];   // 16 rows x 256 k, padded
  __shared__ float    rbV[8 * 16];
  __shared__ int      rbI[8 * 16];
  __shared__ int      fIdx[16];
  __shared__ float    lred[8];

  const int t    = threadIdx.x;
  const int lane = t & 31;
  const int wv   = t >> 5;             // 8 waves
  const int n0   = blockIdx.x * 16;    // first flat row of tile
  const int bb   = n0 >> 10;           // batch index (HW = 1024, tile never crosses)
  const int hw0  = n0 & (HW - 1);

  // ---- stage 1: X tile (strided NCHW gather, coalesced over m) -> LDS f16
  {
    const int d = t;
    const float* g = enc + (size_t)bb * (D_EMB * HW) + (size_t)d * HW + hw0;
    #pragma unroll
    for (int m = 0; m < 16; ++m)
      Xs[m * XS_PITCH + d] = (_Float16)g[m];
  }
  __syncthreads();

  // ---- stage 2: build A fragments (contiguous k = 16*hi + s, matching B), reused 8x
  const int mA = lane & 15;
  const int hi = lane >> 4;
  v16h afrag[8];
  #pragma unroll
  for (int kk = 0; kk < 8; ++kk) {
    v16h a;
    #pragma unroll
    for (int v = 0; v < 8; ++v) {
      const int k0 = kk * 32 + 16 * hi + 2 * v;
      a[2 * v]     = Xs[mA * XS_PITCH + k0];
      a[2 * v + 1] = Xs[mA * XS_PITCH + k0 + 1];
    }
    afrag[kk] = a;
  }

  float bestV[8];
  int   bestI[8];
  #pragma unroll
  for (int r = 0; r < 8; ++r) { bestV[r] = 3.0e38f; bestI[r] = 0; }

  // ---- stage 3: 8 column tiles per wave, 8 WMMA k-steps each
  for (int ti = 0; ti < 8; ++ti) {
    const int ct = wv + ti * 8;                 // column tile 0..63
    const int nn = ct * 16 + mA;                // this lane's code index
    v8f acc = {0.f, 0.f, 0.f, 0.f, 0.f, 0.f, 0.f, 0.f};
    const _Float16* bp = cb16 + (size_t)nn * D_EMB + 16 * hi;  // B 32x16 layout
    #pragma unroll
    for (int kk = 0; kk < 8; ++kk) {
      v16h b = *(const v16h*)(bp + kk * 32);    // 32B contiguous, aligned
      acc = __builtin_amdgcn_wmma_f32_16x16x32_f16(false, afrag[kk], false, b,
                                                   (short)0, acc, false, false);
    }
    const float cn = cnorm[nn];
    #pragma unroll
    for (int r = 0; r < 8; ++r) {               // score = ||c||^2 - 2 x.c
      const float s = cn - 2.0f * acc[r];
      if (s < bestV[r] || (s == bestV[r] && nn < bestI[r])) {
        bestV[r] = s; bestI[r] = nn;
      }
    }
  }

  // ---- stage 4: argmin across the 16 lanes of each half, then across waves
  #pragma unroll
  for (int r = 0; r < 8; ++r) {
    float v = bestV[r]; int i = bestI[r];
    #pragma unroll
    for (int off = 8; off >= 1; off >>= 1) {
      const float ov = __shfl_xor(v, off, 32);
      const int   oi = __shfl_xor(i, off, 32);
      if (ov < v || (ov == v && oi < i)) { v = ov; i = oi; }
    }
    if (mA == 0) {                  // row m = r + 8*hi
      rbV[wv * 16 + r + 8 * hi] = v;
      rbI[wv * 16 + r + 8 * hi] = i;
    }
  }
  __syncthreads();
  if (t < 16) {
    float v = rbV[t]; int i = rbI[t];
    for (int w = 1; w < 8; ++w) {
      const float ov = rbV[w * 16 + t]; const int oi = rbI[w * 16 + t];
      if (ov < v || (ov == v && oi < i)) { v = ov; i = oi; }
    }
    fIdx[t] = i;
    atomicAdd(&counts[i], 1);       // for perplexity
  }
  __syncthreads();

  // ---- stage 5: quantized (NCHW gather of codebook rows) + loss partial
  {
    const int d = t;
    const size_t gbase = (size_t)bb * (D_EMB * HW) + (size_t)d * HW + hw0;
    float ls = 0.f;
    #pragma unroll
    for (int m = 0; m < 16; ++m) {
      const float q  = cb[(size_t)fIdx[m] * D_EMB + d];   // regular-temporal: shared
      const float x  = enc[gbase + m];                    // L2 hit from stage 1
      const float dx = q - x;
      ls += dx * dx;
      __builtin_nontemporal_store(q, out + QOFF + gbase + m);  // streaming output
    }
    #pragma unroll
    for (int off = 16; off >= 1; off >>= 1) ls += __shfl_xor(ls, off, 32);
    if (lane == 0) lred[wv] = ls;
    __syncthreads();
    if (t == 0) {
      float tot = 0.f;
      #pragma unroll
      for (int i = 0; i < 8; ++i) tot += lred[i];
      atomicAdd(lossSum, tot);
    }
  }

  // ---- stage 6: one-hot encodings, coalesced non-temporal streaming stores
  {
    const int c0 = t * 4;
    float* oute = out + EOFF;
    #pragma unroll
    for (int m = 0; m < 16; ++m) {
      const int id = fIdx[m];
      v2f lo, hiv;
      lo.x  = (c0 + 0 == id) ? 1.f : 0.f;
      lo.y  = (c0 + 1 == id) ? 1.f : 0.f;
      hiv.x = (c0 + 2 == id) ? 1.f : 0.f;
      hiv.y = (c0 + 3 == id) ? 1.f : 0.f;
      float* row = oute + (size_t)(n0 + m) * K_CODES + c0;   // 8B aligned
      __builtin_nontemporal_store(lo,  (v2f*)(row));
      __builtin_nontemporal_store(hiv, (v2f*)(row + 2));
    }
  }
}

// ---------------- kernel 3: scalars --------------------------------------
__global__ __launch_bounds__(1024) void vq_final(const int* __restrict__ counts,
                                                 const float* __restrict__ lossSum,
                                                 float* __restrict__ out) {
  __shared__ float red[32];
  const int t = threadIdx.x;
  const float p = (float)counts[t] * (1.0f / (float)N_ROWS);
  float e = p * logf(p + 1e-10f);
  #pragma unroll
  for (int off = 16; off >= 1; off >>= 1) e += __shfl_xor(e, off, 32);
  if ((t & 31) == 0) red[t >> 5] = e;
  __syncthreads();
  if (t == 0) {
    float s = 0.f;
    #pragma unroll
    for (int i = 0; i < 32; ++i) s += red[i];
    out[PERPO] = expf(-s);
    // vq_loss = (1 + BETA) * mean((q - x)^2), BETA = 0.25
    out[0] = 1.25f * (*lossSum) / ((float)N_ROWS * (float)D_EMB);
  }
}

// ---------------- launcher ------------------------------------------------
extern "C" void kernel_launch(void* const* d_in, const int* in_sizes, int n_in,
                              void* d_out, int out_size, void* d_ws, size_t ws_size,
                              hipStream_t stream) {
  const float* enc = (const float*)d_in[0];
  const float* cb  = (const float*)d_in[1];
  float* out = (float*)d_out;

  char* ws = (char*)d_ws;
  _Float16* cb16   = (_Float16*)ws;                 // 1024*256*2 = 512 KB
  float*    cnorm  = (float*)(ws + 524288);         // 4 KB
  int*      counts = (int*)(ws + 528384);           // 4 KB
  float*    lossSm = (float*)(ws + 532480);         // 4 B

  vq_prep <<<K_CODES,      256, 0, stream>>>(cb, cb16, cnorm, counts, lossSm);
  vq_main <<<N_ROWS / 16,  256, 0, stream>>>(enc, cb, cb16, cnorm, counts, lossSm, out);
  vq_final<<<1,           1024, 0, stream>>>(counts, lossSm, out);
}
